// ShortMemory_14328010899739
// MI455X (gfx1250) — matrix operand order repaired and checked by reading, not compiled, not run on hardware
//
#include <hip/hip_runtime.h>

// ---------------- problem constants ----------------
#define B_TOT   8192
#define N_SEQ   48
#define NSC     96     // N_SEQ_CXT
#define NVC     32     // N_VAR_CXT
#define NVT     8      // N_VAR_TIME
#define HDIM    64
#define HORIZON 24
#define SEQ_T   (N_SEQ + HORIZON)   // 72

#define TILE_B   16
#define NTHREADS 256

// padded LDS strides (avoid 64-bank conflicts on stride-96/128 accesses)
#define XKP 34    // halves per (b,s) row of xk   (32 + 2 pad)
#define SCP 97    // floats per b row of score/xl (96 + 1 pad)
#define HSP 130   // halves per b row of [h|s]    (128 + 2 pad)

typedef __attribute__((ext_vector_type(16))) _Float16 v16h;
typedef __attribute__((ext_vector_type(2)))  _Float16 v2h;
typedef __attribute__((ext_vector_type(8)))  float    v8f;

// ---------------- LDS layout (bytes) ----------------
#define OFF_XK  0
#define SZ_XK   (TILE_B * NSC * XKP * 2)       // 104448
#define OFF_XL  (OFF_XK + SZ_XK)
#define SZ_XL   (TILE_B * SCP * 4)             // 6208
#define OFF_SC  (OFF_XL + SZ_XL)
#define SZ_SC   (TILE_B * SCP * 4)             // 6208
#define OFF_Q   (OFF_SC + SZ_SC)
#define SZ_Q    (TILE_B * NVC * 4)             // 2048
#define OFF_HS  (OFF_Q + SZ_Q)
#define SZ_HS   (TILE_B * HSP * 2)             // 4160
#define OFF_S   (OFF_HS + SZ_HS)
#define SZ_S    (TILE_B * HDIM * 4)            // 4096
#define OFF_G   (OFF_S + SZ_S)
#define SZ_G    (TILE_B * 4 * HDIM * 4)        // 16384
#define OFF_Y   (OFF_G + SZ_G)
#define SZ_Y    (TILE_B * N_SEQ * 4)           // 3072
#define OFF_WL  (OFF_Y + SZ_Y)
#define SZ_WL   (NVC * 4)                      // 128
#define OFF_CTX (OFF_WL + SZ_WL)
#define SZ_CTX  (TILE_B * 4)                   // 64
#define SMEM_BYTES (OFF_CTX + SZ_CTX)          // 146816 -> 2 WGs / WGP (320KB)

// ---------------- helpers ----------------
__device__ __forceinline__ float fast_rcp(float x) { return __builtin_amdgcn_rcpf(x); }

__device__ __forceinline__ float fast_tanh(float x) {
  float e = __expf(2.0f * x);                     // v_exp_f32
  return 1.0f - 2.0f * fast_rcp(e + 1.0f);        // exact +/-1 saturation
}
__device__ __forceinline__ float fast_sig(float x) {
  return fast_rcp(1.0f + __expf(-x));
}

__device__ __forceinline__ v8f wmma_f16(v16h a, v16h b, v8f c) {
  return __builtin_amdgcn_wmma_f32_16x16x32_f16(false, a, false, b, (short)0, c, false, false);
}

// A fragment (16x32 f16, row-major in LDS, row stride ld halves)
// layout: lane<16 -> M=lane, halves 0..7 = K0..7, 8..15 = K16..23
//         lane>=16 -> M=lane-16, halves 0..7 = K8..15, 8..15 = K24..31
__device__ __forceinline__ v16h load_A_lds(const _Float16* base, int ld) {
  int lane = threadIdx.x & 31;
  const _Float16* row = base + (lane & 15) * ld;
  int k0 = (lane < 16) ? 0 : 8;
  v16h a;
#pragma unroll
  for (int i = 0; i < 8; ++i) { a[i] = row[k0 + i]; a[8 + i] = row[16 + k0 + i]; }
  return a;
}

// B fragment = W^T tile (32x16), W is (Nout, K) f32 row-major in global.
// B[k][n] = W[n0+n][k0+k]; lane<16: halves 0..7 = K0..7, 8..15 = K16..23; lane>=16: K8..15, K24..31; N = lane&15
__device__ __forceinline__ v16h load_B_wT(const float* W, int ldw, int n0, int k0) {
  int lane = threadIdx.x & 31;
  const float* row = W + (size_t)(n0 + (lane & 15)) * ldw + k0 + ((lane < 16) ? 0 : 8);
  v16h b;
#pragma unroll
  for (int i = 0; i < 8; ++i) { b[i] = (_Float16)row[i]; b[8 + i] = (_Float16)row[16 + i]; }
  return b;
}

// C fragment = per-column bias broadcast (f32 16x16 C/D layout)
__device__ __forceinline__ v8f bcast_col(const float* bv, int n0) {
  int lane = threadIdx.x & 31;
  float val = bv[n0 + (lane & 15)];
  v8f c;
#pragma unroll
  for (int r = 0; r < 8; ++r) c[r] = val;
  return c;
}
__device__ __forceinline__ v8f bcast_col2(const float* b0v, const float* b1v, int n0) {
  int lane = threadIdx.x & 31;
  float val = b0v[n0 + (lane & 15)] + b1v[n0 + (lane & 15)];
  v8f c;
#pragma unroll
  for (int r = 0; r < 8; ++r) c[r] = val;
  return c;
}

__device__ __forceinline__ void store_D_f32(float* base, int ld, v8f d) {
  int lane = threadIdx.x & 31;
  int col = lane & 15;
  int r0 = (lane < 16) ? 0 : 8;
#pragma unroll
  for (int r = 0; r < 8; ++r) base[(r0 + r) * ld + col] = d[r];
}

// ---------------- main fused kernel: one 16-batch tile per workgroup ----------------
__global__ void __launch_bounds__(NTHREADS)
lstm_attn_kernel(const float* __restrict__ x,  const float* __restrict__ y,
                 const float* __restrict__ Wq, const float* __restrict__ bq,
                 const float* __restrict__ Wk, const float* __restrict__ bk,
                 const float* __restrict__ vvec,
                 const float* __restrict__ Wl, const float* __restrict__ bl,
                 const float* __restrict__ Wih,const float* __restrict__ Whh,
                 const float* __restrict__ bih,const float* __restrict__ bhh,
                 const float* __restrict__ Wf, const float* __restrict__ bf,
                 float* __restrict__ out)
{
  extern __shared__ char sm[];
  _Float16* xk   = (_Float16*)(sm + OFF_XK);
  float*    xl   = (float*)(sm + OFF_XL);
  float*    score= (float*)(sm + OFF_SC);
  float*    qb   = (float*)(sm + OFF_Q);
  _Float16* hs   = (_Float16*)(sm + OFF_HS);
  float*    sbuf = (float*)(sm + OFF_S);
  float*    gates= (float*)(sm + OFF_G);
  float*    ytile= (float*)(sm + OFF_Y);
  float*    wl_s = (float*)(sm + OFF_WL);
  float*    ctxl = (float*)(sm + OFF_CTX);

  const int tid  = threadIdx.x;
  const int lane = tid & 31;
  const int warp = tid >> 5;
  const int b0   = blockIdx.x * TILE_B;

  // ---- stage small data / init state ----
  if (tid < NVC) wl_s[tid] = Wl[tid];
  for (int i = tid; i < TILE_B * N_SEQ; i += NTHREADS) {
    int b = i / N_SEQ, tt = i % N_SEQ;
    ytile[b * N_SEQ + tt] = y[(size_t)(b0 + b) * N_SEQ + tt];
  }
  for (int i = tid; i < TILE_B * HSP; i += NTHREADS) hs[i] = (_Float16)0.0f;
  for (int i = tid; i < TILE_B * HDIM; i += NTHREADS) sbuf[i] = 0.0f;
  __syncthreads();

  // ---- register-resident weight fragments ----
  v16h wk_b[2];
  wk_b[0] = load_B_wT(Wk, NVC, 0, 0);
  wk_b[1] = load_B_wT(Wk, NVC, 16, 0);
  v8f  bk_c[2]; bk_c[0] = bcast_col(bk, 0); bk_c[1] = bcast_col(bk, 16);

  v16h wq_b[2][4];
  v8f  qb_c[2];
  if (warp == 0) {
#pragma unroll
    for (int nt = 0; nt < 2; ++nt) {
      qb_c[nt] = bcast_col(bq, nt * 16);
#pragma unroll
      for (int kc = 0; kc < 4; ++kc) wq_b[nt][kc] = load_B_wT(Wq, 2 * HDIM, nt * 16, kc * 32);
    }
  }

  v16h whh_b[2][2];
  v8f  gb_c[2];
  float wih0[2], wih1[2];
#pragma unroll
  for (int nti = 0; nti < 2; ++nti) {
    int n0 = (2 * warp + nti) * 16;
    gb_c[nti]       = bcast_col2(bih, bhh, n0);
    whh_b[nti][0]   = load_B_wT(Whh, HDIM, n0, 0);
    whh_b[nti][1]   = load_B_wT(Whh, HDIM, n0, 32);
    int col = n0 + (lane & 15);
    wih0[nti] = Wih[2 * col];
    wih1[nti] = Wih[2 * col + 1];
  }

  float vreg[NVC];
#pragma unroll
  for (int c = 0; c < NVC; ++c) vreg[c] = vvec[c];
  const float bl0 = bl[0];

  // ---- setup: xk = x@Wk^T + bk (WMMA, f16 into LDS), xl = x . Wl ----
  for (int mt = warp; mt < (TILE_B * NSC) / 16; mt += 8) {
    int bL = mt / (NSC / 16);
    int s0 = (mt % (NSC / 16)) * 16;
    const float* xr = x + ((size_t)(b0 + bL) * NSC + s0 + (lane & 15)) * NVC;
    int k0h = (lane < 16) ? 0 : 8;
    v16h a;
#pragma unroll
    for (int i = 0; i < 8; ++i) { a[i] = (_Float16)xr[k0h + i]; a[8 + i] = (_Float16)xr[16 + k0h + i]; }
#pragma unroll
    for (int nt = 0; nt < 2; ++nt) {
      v8f d = wmma_f16(a, wk_b[nt], bk_c[nt]);
      _Float16* dst = xk + (size_t)(bL * NSC + s0) * XKP + nt * 16 + (lane & 15);
      int r0 = (lane < 16) ? 0 : 8;
#pragma unroll
      for (int r = 0; r < 8; ++r) dst[(size_t)(r0 + r) * XKP] = (_Float16)d[r];
    }
  }
  for (int r = tid; r < TILE_B * NSC; r += NTHREADS) {
    int bL = r / NSC, s = r % NSC;
    const float* xr = x + ((size_t)(b0 + bL) * NSC + s) * NVC;
    float acc = 0.0f;
#pragma unroll
    for (int c = 0; c < NVC; ++c) acc += xr[c] * wl_s[c];
    xl[bL * SCP + s] = acc;
  }
  __syncthreads();

  // ---- 48-step attention-LSTM recurrence (all on-chip) ----
  for (int st = 0; st < N_SEQ; ++st) {
    // (1) q = [h,s] @ Wq^T + bq   (wave 0, 8 WMMAs)
    if (warp == 0) {
      v16h aq[4];
#pragma unroll
      for (int kc = 0; kc < 4; ++kc) aq[kc] = load_A_lds(hs + kc * 32, HSP);
#pragma unroll
      for (int nt = 0; nt < 2; ++nt) {
        v8f acc = qb_c[nt];
#pragma unroll
        for (int kc = 0; kc < 4; ++kc) acc = wmma_f16(aq[kc], wq_b[nt][kc], acc);
        store_D_f32(qb + nt * 16, NVC, acc);
      }
    }
    __syncthreads();

    // (2) scores = tanh(q + xk) . v   (all 256 threads; 6 positions each)
    {
      int b = tid >> 4;
      int sl = tid & 15;
      float qr[NVC];
      const float* qrow = qb + b * NVC;
#pragma unroll
      for (int c = 0; c < NVC; ++c) qr[c] = qrow[c];
#pragma unroll
      for (int ss = 0; ss < 6; ++ss) {
        int s = sl + 16 * ss;
        const v2h* xkr = (const v2h*)(xk + (size_t)(b * NSC + s) * XKP);
        float acc = 0.0f;
#pragma unroll
        for (int c2 = 0; c2 < 16; ++c2) {
          v2h p = xkr[c2];
          int c = c2 * 2;
          acc += fast_tanh(qr[c]     + (float)p[0]) * vreg[c];
          acc += fast_tanh(qr[c + 1] + (float)p[1]) * vreg[c + 1];
        }
        score[b * SCP + s] = acc;
      }
    }
    __syncthreads();

    // (3) softmax folded into scalar context: ctxl = (Σ e·xl)/(Σ e) + bl
    if (tid < TILE_B) {
      float* sc = score + tid * SCP;
      const float* xlr = xl + tid * SCP;
      float mx = sc[0];
      for (int s = 1; s < NSC; ++s) mx = fmaxf(mx, sc[s]);
      float sum = 0.0f, cacc = 0.0f;
      for (int s = 0; s < NSC; ++s) {
        float e = __expf(sc[s] - mx);
        sum += e; cacc += e * xlr[s];
      }
      ctxl[tid] = cacc * fast_rcp(sum) + bl0;
    }
    __syncthreads();

    // (4) gates = h @ Whh^T + (bih+bhh) + Wih[:,0]*ctxl + Wih[:,1]*y_t   (8 waves x 4 WMMAs)
    {
      v16h ah0 = load_A_lds(hs, HSP);
      v16h ah1 = load_A_lds(hs + 32, HSP);
      int r0 = (lane < 16) ? 0 : 8;
#pragma unroll
      for (int nti = 0; nti < 2; ++nti) {
        int n0 = (2 * warp + nti) * 16;
        v8f acc = gb_c[nti];
        acc = wmma_f16(ah0, whh_b[nti][0], acc);
        acc = wmma_f16(ah1, whh_b[nti][1], acc);
        int col = n0 + (lane & 15);
#pragma unroll
        for (int r = 0; r < 8; ++r) {
          int b = r0 + r;
          gates[b * (4 * HDIM) + col] =
              acc[r] + wih0[nti] * ctxl[b] + wih1[nti] * ytile[b * N_SEQ + st];
        }
      }
    }
    __syncthreads();

    // (5) LSTM pointwise update; h,s -> hs (f16), s also kept f32
    for (int i = tid; i < TILE_B * HDIM; i += NTHREADS) {
      int b = i >> 6, j = i & (HDIM - 1);
      const float* g = gates + b * (4 * HDIM);
      float gi = g[j], gf = g[HDIM + j], gg = g[2 * HDIM + j], go = g[3 * HDIM + j];
      float snew = fast_sig(gf) * sbuf[i] + fast_sig(gi) * fast_tanh(gg);
      float hnew = fast_sig(go) * fast_tanh(snew);
      sbuf[i] = snew;
      hs[b * HSP + j]        = (_Float16)hnew;
      hs[b * HSP + HDIM + j] = (_Float16)snew;
    }
    __syncthreads();
  }

  // ---- head: y_pred = h_n @ Wf^T + bf ----
  for (int i = tid; i < TILE_B * HORIZON; i += NTHREADS) {
    int b = i / HORIZON, k = i % HORIZON;
    float acc = bf[k];
    const _Float16* hrow = hs + b * HSP;
    const float* wfr = Wf + (size_t)k * HDIM;
#pragma unroll
    for (int j = 0; j < HDIM; ++j) acc += (float)hrow[j] * wfr[j];
    out[(size_t)(b0 + b) * HORIZON + k] = acc;
  }
}

// ---------------- time-feature MLP, accumulated into y_pred ----------------
__global__ void __launch_bounds__(256)
ft_add_kernel(const float* __restrict__ t,
              const float* __restrict__ Wt1, const float* __restrict__ bt1,
              const float* __restrict__ Wt2, const float* __restrict__ bt2,
              float* __restrict__ out)
{
  int idx = blockIdx.x * blockDim.x + threadIdx.x;
  if (idx >= B_TOT * HORIZON) return;
  int b = idx / HORIZON, k = idx % HORIZON;
  const float* tt = t + ((size_t)b * SEQ_T + (SEQ_T - HORIZON) + k) * NVT;
  float tr[NVT];
#pragma unroll
  for (int c = 0; c < NVT; ++c) tr[c] = tt[c];
  float acc = bt2[0];
#pragma unroll
  for (int j = 0; j < 16; ++j) {
    float h = bt1[j];
    const float* w = Wt1 + j * NVT;
#pragma unroll
    for (int c = 0; c < NVT; ++c) h += tr[c] * w[c];
    acc += fmaxf(h, 0.0f) * Wt2[j];
  }
  out[idx] += acc;
}

// ---------------- y_encoded / y_decoded = zeros ----------------
__global__ void __launch_bounds__(256)
zero_tail_kernel(float4* __restrict__ outTail, int n4)
{
  int i = blockIdx.x * blockDim.x + threadIdx.x;
  if (i < n4) outTail[i] = make_float4(0.f, 0.f, 0.f, 0.f);
}

// ---------------- host launcher ----------------
extern "C" void kernel_launch(void* const* d_in, const int* in_sizes, int n_in,
                              void* d_out, int out_size, void* d_ws, size_t ws_size,
                              hipStream_t stream) {
  const float* x   = (const float*)d_in[0];
  const float* y   = (const float*)d_in[1];
  const float* t   = (const float*)d_in[2];
  const float* Wq  = (const float*)d_in[3];
  const float* bq  = (const float*)d_in[4];
  const float* Wk  = (const float*)d_in[5];
  const float* bk  = (const float*)d_in[6];
  const float* v   = (const float*)d_in[7];
  const float* Wl  = (const float*)d_in[8];
  const float* bl  = (const float*)d_in[9];
  const float* Wih = (const float*)d_in[10];
  const float* Whh = (const float*)d_in[11];
  const float* bih = (const float*)d_in[12];
  const float* bhh = (const float*)d_in[13];
  const float* Wt1 = (const float*)d_in[14];
  const float* bt1 = (const float*)d_in[15];
  const float* Wt2 = (const float*)d_in[16];
  const float* bt2 = (const float*)d_in[17];
  const float* Wf  = (const float*)d_in[18];
  const float* bf  = (const float*)d_in[19];
  float* out = (float*)d_out;

  (void)hipFuncSetAttribute((const void*)lstm_attn_kernel,
                            hipFuncAttributeMaxDynamicSharedMemorySize, SMEM_BYTES);

  lstm_attn_kernel<<<B_TOT / TILE_B, NTHREADS, SMEM_BYTES, stream>>>(
      x, y, Wq, bq, Wk, bk, v, Wl, bl, Wih, Whh, bih, bhh, Wf, bf, out);

  ft_add_kernel<<<(B_TOT * HORIZON + 255) / 256, 256, 0, stream>>>(
      t, Wt1, bt1, Wt2, bt2, out);

  int tail4 = (B_TOT * 2 * N_SEQ) / 4;
  zero_tail_kernel<<<(tail4 + 255) / 256, 256, 0, stream>>>(
      (float4*)(out + (size_t)B_TOT * HORIZON), tail4);
}
